// RGCNLayer_15006615732837
// MI455X (gfx1250) — matrix operand order repaired and checked by reading, not compiled, not run on hardware
//
#include <hip/hip_runtime.h>

typedef __attribute__((ext_vector_type(16))) _Float16 v16h;
typedef __attribute__((ext_vector_type(8)))  _Float16 v8h;
typedef __attribute__((ext_vector_type(2)))  _Float16 v2h;
typedef __attribute__((ext_vector_type(8)))  float    v8f;

#define N_NODES 50000
#define N_EDGES 500000
#define DIM     128
#define N_REL   8
#define BLK_M   64
#define WPAD    136   // padded K-stride (halves) for transposed W: 272B rows

// ---------------------------------------------------------------------------
// Kernel 1: zero the output accumulator (harness poisons d_out with 0xAA).
// ---------------------------------------------------------------------------
__global__ __launch_bounds__(256)
void rgcn_zero(float* __restrict__ out, int n)
{
    int i = blockIdx.x * 256 + threadIdx.x;
    if (i < n) out[i] = 0.0f;
}

// ---------------------------------------------------------------------------
// Kernel 2: xw[r] = x @ W[r], f16 WMMA / f32 accumulate.
// Block = 256 threads = 8 waves; block tile 64 x 128.
//   wave w: M-subtile (w>>1)*16, N-half (w&1)*64 -> 4x 16x16 WMMA tiles.
// LDS: sX row-major [m][k] (A fragments = 2x ds_load_b128),
//      sWt transposed [n][k], padded rows (B fragments = 2x ds_load_b128).
// ---------------------------------------------------------------------------
__global__ __launch_bounds__(256)
void rgcn_gemm(const float* __restrict__ x,
               const float* __restrict__ w,
               float* __restrict__ xw)
{
    __shared__ _Float16 sX[BLK_M][DIM];     // 16 KB
    __shared__ _Float16 sWt[DIM][WPAD];     // 34 KB, sWt[n][k] = W[k][n]

    const int rel  = blockIdx.y;
    const int row0 = blockIdx.x * BLK_M;
    const int tid  = threadIdx.x;

    // ---- Stage W[rel] transposed: read coalesced (consecutive n), write
    // packed k-pairs (ds_store_b32), 8192 pairs / 256 threads = 32 iters.
    const float* wr = w + (size_t)rel * DIM * DIM;
    #pragma unroll
    for (int i = tid; i < (DIM / 2) * DIM; i += 256) {
        const int kp = i >> 7;        // k pair index 0..63
        const int n  = i & 127;
        const float f0 = wr[(2 * kp) * DIM + n];
        const float f1 = wr[(2 * kp + 1) * DIM + n];
        v2h h; h[0] = (_Float16)f0; h[1] = (_Float16)f1;
        *(v2h*)&sWt[n][2 * kp] = h;
    }

    // ---- Stage x tile (float2 -> packed v2h), rows past N_NODES -> 0.
    #pragma unroll
    for (int i = tid; i < BLK_M * (DIM / 2); i += 256) {
        const int r = i >> 6;
        const int c = (i & 63) * 2;
        const int gr = row0 + r;
        float f0 = 0.0f, f1 = 0.0f;
        if (gr < N_NODES) {
            const float2 f = *(const float2*)&x[(size_t)gr * DIM + c];
            f0 = f.x; f1 = f.y;
        }
        v2h h; h[0] = (_Float16)f0; h[1] = (_Float16)f1;
        *(v2h*)&sX[r][c] = h;
    }
    __syncthreads();

    const int wave = tid >> 5;
    const int lane = tid & 31;
    const int mt   = wave >> 1;      // M subtile 0..3
    const int nh   = wave & 1;       // N half 0..1
    const int lm   = lane & 15;
    const int lhi  = lane >> 4;      // 0 or 1

    v8f acc[4];
    #pragma unroll
    for (int t = 0; t < 4; ++t) acc[t] = (v8f){0,0,0,0,0,0,0,0};

    const int m = mt * 16 + lm;

    #pragma unroll
    for (int kc = 0; kc < 4; ++kc) {
        const int kb = kc * 32;

        // A fragment: VGPRs 0-3 = halves [kofA, kofA+7], VGPRs 4-7 = [+16, +23]
        const int kofA = kb + lhi * 8;
        const v8h alo = *(const v8h*)&sX[m][kofA];
        const v8h ahi = *(const v8h*)&sX[m][kofA + 16];
        const v16h a = __builtin_shufflevector(alo, ahi,
            0, 1, 2, 3, 4, 5, 6, 7, 8, 9, 10, 11, 12, 13, 14, 15);

        const int kofB = kb + lhi * 8;
        #pragma unroll
        for (int nt = 0; nt < 4; ++nt) {
            const int n = nh * 64 + nt * 16 + lm;
            // B fragment: halves k = [kofB, kofB+7], [kofB+16, kofB+23] at fixed column n
            const v8h blo = *(const v8h*)&sWt[n][kofB];
            const v8h bhi = *(const v8h*)&sWt[n][kofB + 16];
            const v16h b = __builtin_shufflevector(blo, bhi,
                0, 1, 2, 3, 4, 5, 6, 7, 8, 9, 10, 11, 12, 13, 14, 15);

            acc[nt] = __builtin_amdgcn_wmma_f32_16x16x32_f16(
                false, a, false, b, (short)0, acc[nt], false, false);
        }
    }

    // ---- Store C (lane group lhi -> rows +8, VGPR i -> row+i, N=lm)
    float* dst = xw + (size_t)rel * N_NODES * DIM;
    #pragma unroll
    for (int nt = 0; nt < 4; ++nt) {
        const int col = nh * 64 + nt * 16 + lm;
        #pragma unroll
        for (int i = 0; i < 8; ++i) {
            const int row = row0 + mt * 16 + lhi * 8 + i;
            if (row < N_NODES)
                dst[(size_t)row * DIM + col] = acc[nt][i];
        }
    }
}

// ---------------------------------------------------------------------------
// Kernel 3: per-edge gather + bias + relu + scatter-add (atomics).
// One wave per edge; lane handles 4 consecutive floats (float4 = 512B/edge).
// ---------------------------------------------------------------------------
__global__ __launch_bounds__(256)
void rgcn_edges(const float* __restrict__ xw,
                const float* __restrict__ bias,
                const int*   __restrict__ edge_index,  // [2, E] flat
                const int*   __restrict__ edge_type,   // [E]
                float*       __restrict__ out)
{
    const int e = blockIdx.x * 8 + (threadIdx.x >> 5);
    if (e >= N_EDGES) return;
    const int lane = threadIdx.x & 31;

    const int s = edge_index[e];
    const int d = edge_index[N_EDGES + e];
    const int r = edge_type[e];

    const float4 v  = *(const float4*)(xw + (((size_t)r * N_NODES + s) * DIM) + lane * 4);
    const float4 bv = *(const float4*)(bias + lane * 4);

    const float m0 = fmaxf(v.x + bv.x, 0.0f);
    const float m1 = fmaxf(v.y + bv.y, 0.0f);
    const float m2 = fmaxf(v.z + bv.z, 0.0f);
    const float m3 = fmaxf(v.w + bv.w, 0.0f);

    float* o = out + (size_t)d * DIM + lane * 4;
    __hip_atomic_fetch_add(o + 0, m0, __ATOMIC_RELAXED, __HIP_MEMORY_SCOPE_AGENT);
    __hip_atomic_fetch_add(o + 1, m1, __ATOMIC_RELAXED, __HIP_MEMORY_SCOPE_AGENT);
    __hip_atomic_fetch_add(o + 2, m2, __ATOMIC_RELAXED, __HIP_MEMORY_SCOPE_AGENT);
    __hip_atomic_fetch_add(o + 3, m3, __ATOMIC_RELAXED, __HIP_MEMORY_SCOPE_AGENT);
}

// ---------------------------------------------------------------------------
extern "C" void kernel_launch(void* const* d_in, const int* in_sizes, int n_in,
                              void* d_out, int out_size, void* d_ws, size_t ws_size,
                              hipStream_t stream)
{
    const float* x          = (const float*)d_in[0];  // [50000,128]
    const float* weight     = (const float*)d_in[1];  // [8,128,128]
    const float* bias       = (const float*)d_in[2];  // [128]
    const int*   edge_index = (const int*)d_in[3];    // [2,500000]
    const int*   edge_type  = (const int*)d_in[4];    // [500000]
    float*       out        = (float*)d_out;          // [50000,128]
    float*       xw         = (float*)d_ws;           // [8,50000,128] f32 = 204.8 MB

    // 1) zero output accumulator
    {
        const int n = N_NODES * DIM;
        rgcn_zero<<<(n + 255) / 256, 256, 0, stream>>>(out, n);
    }
    // 2) xw[r] = x @ W[r]  (WMMA f16 -> f32)
    {
        dim3 grid((N_NODES + BLK_M - 1) / BLK_M, N_REL);
        rgcn_gemm<<<grid, 256, 0, stream>>>(x, weight, xw);
    }
    // 3) gather + relu + scatter-add
    {
        rgcn_edges<<<(N_EDGES + 7) / 8, 256, 0, stream>>>(xw, bias, edge_index, edge_type, out);
    }
}
